// MOE_Resnet18_Top_1_20272245637144
// MI455X (gfx1250) — matrix operand order repaired and checked by reading, not compiled, not run on hardware
//
#include <hip/hip_runtime.h>
#include <hip/hip_bf16.h>

// ---------------------------------------------------------------------------
// MoE (top-1) ResNet18, MI455X / gfx1250.
// Convs = implicit GEMM on v_wmma_f32_16x16x32_bf16 (fp32 accumulate).
//  - weights:     bf16 [E][Cout][tap][Cin_pad], Cin_pad = max(32, Cin)
//  - activations: bf16 NHWC with a zero spatial halo: [(H+2)*(W+2)][C]
//  - register blocking: each wave computes MSUB x NSUB 16x16 WMMA tiles,
//    reusing A across NSUB and B across MSUB => ~2 b128 loads per wmma.
//  - fragment loads/stores use explicit address_space(1) casts so the
//    backend emits global_load_b128 (not flat_load_b128 + DScnt coupling).
// ---------------------------------------------------------------------------

typedef __attribute__((ext_vector_type(16))) __bf16    v16bf;
typedef __attribute__((ext_vector_type(8)))  float     v8f;
typedef __attribute__((ext_vector_type(4)))  unsigned  u32x4;

#define BN_EPS 1e-5f

union BFrag {
    v16bf bf;
    u32x4 v[2];
};
union OFrag {
    u32x4          v;
    unsigned short us[8];
};

typedef const __attribute__((address_space(1))) u32x4* gcu32x4p;
typedef       __attribute__((address_space(1))) u32x4* gu32x4p;

__device__ __forceinline__ u32x4 gld(const unsigned short* p) {
    return *(gcu32x4p)p;                    // force GLOBAL addrspace load
}
__device__ __forceinline__ void gst(unsigned short* p, u32x4 v) {
    *(gu32x4p)p = v;                        // force GLOBAL addrspace store
}

__device__ __forceinline__ unsigned short f2bf(float f) {
    unsigned u = __float_as_uint(f);
    unsigned r = u + 0x7FFFu + ((u >> 16) & 1u);   // round-to-nearest-even
    return (unsigned short)(r >> 16);
}
__device__ __forceinline__ float bf2f(unsigned short h) {
    return __uint_as_float((unsigned)h << 16);
}

// ---------------------------------------------------------------------------
// Zero a workspace region (16B granules).
// ---------------------------------------------------------------------------
__global__ void zero_kernel(u32x4* __restrict__ p, long n16) {
    long i = (long)blockIdx.x * blockDim.x + threadIdx.x;
    if (i < n16) {
        u32x4 z = {0u, 0u, 0u, 0u};
        p[i] = z;
    }
}

// ---------------------------------------------------------------------------
// Weights: src [E*Cout][Cin][k2] fp32 -> dst [E*Cout][k2][Cinp] bf16 (zero pad)
// ---------------------------------------------------------------------------
__global__ void convert_w_kernel(const float* __restrict__ src,
                                 unsigned short* __restrict__ dst,
                                 long total, int Cin, int Cinp, int k2) {
    long i = (long)blockIdx.x * blockDim.x + threadIdx.x;
    if (i >= total) return;
    int  ci = (int)(i % Cinp);
    long r  = i / Cinp;
    int  g  = (int)(r % k2);
    long co = r / k2;                       // spans E*Cout
    float v = 0.f;
    if (ci < Cin) v = src[(co * Cin + ci) * k2 + g];
    dst[i] = f2bf(v);
}

// ---------------------------------------------------------------------------
// Stem input: x [128][3][32][32] fp32 NCHW -> bf16 NHWC 34x34x32, zero halo
// and zero channels 3..31.
// ---------------------------------------------------------------------------
__global__ void convert_x_kernel(const float* __restrict__ x,
                                 unsigned short* __restrict__ dst) {
    long i = (long)blockIdx.x * blockDim.x + threadIdx.x;
    if (i >= 128L * 34 * 34 * 32) return;
    int  c    = (int)(i & 31);
    long t    = i >> 5;
    int  cell = (int)(t % (34 * 34));
    int  s    = (int)(t / (34 * 34));
    int  r    = cell / 34;
    int  cl   = cell - r * 34;
    float v = 0.f;
    if (r >= 1 && r <= 32 && cl >= 1 && cl <= 32 && c < 3)
        v = x[(size_t)s * 3072 + (size_t)c * 1024 + (r - 1) * 32 + (cl - 1)];
    dst[i] = f2bf(v);
}

// ---------------------------------------------------------------------------
// Gate: one wave32 per sample; argmax with first-max tie-break.
// ---------------------------------------------------------------------------
__global__ void gate_kernel(const float* __restrict__ x,
                            const float* __restrict__ gw,
                            const float* __restrict__ gb,
                            int* __restrict__ top1) {
    int b = blockIdx.x;
    int lane = threadIdx.x;
    const float* xb = x + (size_t)b * 3072;
    float a0 = 0.f, a1 = 0.f, a2 = 0.f, a3 = 0.f;
    for (int i = lane; i < 3072; i += 32) {
        float v = xb[i];
        a0 += v * gw[i];
        a1 += v * gw[3072 + i];
        a2 += v * gw[2 * 3072 + i];
        a3 += v * gw[3 * 3072 + i];
    }
    for (int off = 16; off > 0; off >>= 1) {
        a0 += __shfl_down(a0, off, 32);
        a1 += __shfl_down(a1, off, 32);
        a2 += __shfl_down(a2, off, 32);
        a3 += __shfl_down(a3, off, 32);
    }
    if (lane == 0) {
        float l[4] = {a0 + gb[0], a1 + gb[1], a2 + gb[2], a3 + gb[3]};
        int best = 0;
        float bv = l[0];
        for (int e = 1; e < 4; ++e)
            if (l[e] > bv) { bv = l[e]; best = e; }
        top1[b] = best;
    }
}

// ---------------------------------------------------------------------------
// Fused conv + BN (+bf16 residual) (+ReLU), implicit GEMM on WMMA bf16.
//   Wave tile = (16*MSUB) out-channels x (16*NSUB) pixels.
//   8 waves/block arranged (8>>wavesNLog) x (1<<wavesNLog).
//   Tiling is exact for every layer => no bounds checks in the hot loop.
// ---------------------------------------------------------------------------
template <int MSUB, int NSUB>
__global__ __launch_bounds__(256) void conv_bn_kernel(
    const unsigned short* __restrict__ wB,     // bf16 [E][Cout][Ktot]
    const float* __restrict__ bnBeta, const float* __restrict__ bnGamma,
    const float* __restrict__ bnMean, const float* __restrict__ bnVar,
    const unsigned short* __restrict__ inBase, int inStride,   // bf16 NHWC+halo
    const unsigned short* __restrict__ resBase,                // geom == out
    unsigned short* __restrict__ outBase, int outStride,
    const int* __restrict__ top1,
    int Cinp, int WpIn,
    int Cout, int Hout, int Wout, int WpOut,
    int k2, int stride, int hoff,              // hoff = 1 - pad
    int wavesNLog, int relu)
{
    const int b    = blockIdx.z;
    const int e    = top1[b];
    const int Ktot = k2 * Cinp;
    const int Npix = Hout * Wout;

    const int waveId = threadIdx.x >> 5;
    const int lane   = threadIdx.x & 31;
    const int wavesN = 1 << wavesNLog;
    const int wavesM = 8 >> wavesNLog;
    const int wm = waveId >> wavesNLog;
    const int wn = waveId & (wavesN - 1);
    const int mBase = blockIdx.x * (16 * MSUB * wavesM) + wm * (16 * MSUB);
    const int pBase = blockIdx.y * (16 * NSUB * wavesN) + wn * (16 * NSUB);

    const int laneLo = lane & 15;
    const int hi     = lane >> 4;          // half-wave select (WMMA layout)

    const unsigned short* inS = inBase + (size_t)b * inStride;

    const unsigned short* wRow[MSUB];
    #pragma unroll
    for (int s = 0; s < MSUB; ++s)
        wRow[s] = wB + ((size_t)e * Cout + (mBase + 16 * s + laneLo)) * Ktot
                     + hi * 8;

    int ohv[NSUB], owv[NSUB];
    #pragma unroll
    for (int t = 0; t < NSUB; ++t) {
        int p = pBase + 16 * t + laneLo;
        ohv[t] = p / Wout;
        owv[t] = p - ohv[t] * Wout;
    }

    v8f acc[MSUB][NSUB];
    #pragma unroll
    for (int s = 0; s < MSUB; ++s)
        #pragma unroll
        for (int t = 0; t < NSUB; ++t)
            acc[s][t] = (v8f){0.f, 0.f, 0.f, 0.f, 0.f, 0.f, 0.f, 0.f};

    for (int g = 0; g < k2; ++g) {
        const int kh = (k2 == 9) ? (g / 3) : 0;
        const int kw = g - kh * 3;
        const unsigned short* bRow[NSUB];
        #pragma unroll
        for (int t = 0; t < NSUB; ++t) {
            const int ihh = ohv[t] * stride + kh + hoff;   // halo: always valid
            const int iwh = owv[t] * stride + kw + hoff;
            bRow[t] = inS + ((size_t)ihh * WpIn + iwh) * Cinp + hi * 8;
        }

        for (int ci0 = 0; ci0 < Cinp; ci0 += 32) {
            BFrag af[MSUB], bf[NSUB];
            #pragma unroll
            for (int s = 0; s < MSUB; ++s) {
                af[s].v[0] = gld(wRow[s]);
                af[s].v[1] = gld(wRow[s] + 16);
                wRow[s] += 32;             // contiguous across taps too
            }
            #pragma unroll
            for (int t = 0; t < NSUB; ++t) {
                bf[t].v[0] = gld(bRow[t]);
                bf[t].v[1] = gld(bRow[t] + 16);
                bRow[t] += 32;
            }
            #pragma unroll
            for (int s = 0; s < MSUB; ++s)
                #pragma unroll
                for (int t = 0; t < NSUB; ++t)
                    acc[s][t] = __builtin_amdgcn_wmma_f32_16x16x32_bf16(
                        false, af[s].bf, false, bf[t].bf,
                        (short)0, acc[s][t], false, false);
        }
    }

    // ---- epilogue: BN + residual + ReLU, packed bf16 NHWC stores ----
    const int eC = e * Cout;
    #pragma unroll
    for (int s = 0; s < MSUB; ++s) {
        const int c0 = mBase + 16 * s + hi * 8;   // 8 contiguous out channels
        float sc[8], sh[8];
        #pragma unroll
        for (int r = 0; r < 8; ++r) {
            int c = c0 + r;
            float inv = rsqrtf(bnVar[eC + c] + BN_EPS);
            sc[r] = bnGamma[eC + c] * inv;
            sh[r] = bnBeta[eC + c] - bnMean[eC + c] * sc[r];
        }
        #pragma unroll
        for (int t = 0; t < NSUB; ++t) {
            const int p = pBase + 16 * t + laneLo;
            if (p >= Npix) continue;               // tiling exact; kept cheap
            const size_t po =
                ((size_t)(ohv[t] + 1) * WpOut + (owv[t] + 1)) * Cout + c0;

            float rv[8] = {0.f, 0.f, 0.f, 0.f, 0.f, 0.f, 0.f, 0.f};
            if (resBase) {
                OFrag rf;
                rf.v = gld(resBase + (size_t)b * outStride + po);
                #pragma unroll
                for (int r = 0; r < 8; ++r) rv[r] = bf2f(rf.us[r]);
            }
            OFrag of;
            #pragma unroll
            for (int r = 0; r < 8; ++r) {
                float o = acc[s][t][r] * sc[r] + sh[r] + rv[r];
                if (relu) o = fmaxf(o, 0.f);
                of.us[r] = f2bf(o);
            }
            gst(outBase + (size_t)b * outStride + po, of.v);
        }
    }
}

// ---------------------------------------------------------------------------
// Head: global avg pool over 4x4 (bf16 NHWC 6x6x512 halo) + FC(512->10).
// ---------------------------------------------------------------------------
__global__ __launch_bounds__(256) void head_kernel(
    const unsigned short* __restrict__ act, int actStride,
    const float* __restrict__ fcw, const float* __restrict__ fcb,
    const int* __restrict__ top1, float* __restrict__ out)
{
    __shared__ float pooled[512];
    const int b = blockIdx.x;
    const int t = threadIdx.x;
    const unsigned short* a = act + (size_t)b * actStride;
    for (int c = t; c < 512; c += 256) {
        float s = 0.f;
        #pragma unroll
        for (int r = 0; r < 4; ++r)
            #pragma unroll
            for (int cl = 0; cl < 4; ++cl)
                s += bf2f(a[((size_t)(r + 1) * 6 + (cl + 1)) * 512 + c]);
        pooled[c] = s * (1.f / 16.f);
    }
    __syncthreads();
    const int e = top1[b];
    if (t < 10) {
        const float* w = fcw + (size_t)e * 10 * 512 + (size_t)t * 512;
        float s = fcb[e * 10 + t];
        for (int i = 0; i < 512; ++i) s += w[i] * pooled[i];
        out[b * 10 + t] = s;
    }
}

// ---------------------------------------------------------------------------
// Host orchestration. Input index map (jax pytree: dicts sorted, lists order):
//  0 x | 1 gate_w | 2 gate_b | 3..6 bn1 | 7 conv1 | 8 fc_b | 9 fc_w
//  layer1: b0 bn1@10 bn2@14 c1@18 c2@19 ; b1 bn1@20 bn2@24 c1@28 c2@29
//  layer2: b0 bn1@30 bn2@34 bnd@38 c1@42 c2@43 cd@44 ; b1 bn1@45 bn2@49 c1@53 c2@54
//  layer3: b0 bn1@55 bn2@59 bnd@63 c1@67 c2@68 cd@69 ; b1 bn1@70 bn2@74 c1@78 c2@79
//  layer4: b0 bn1@80 bn2@84 bnd@88 c1@92 c2@93 cd@94 ; b1 bn1@95 bn2@99 c1@103 c2@104
// ---------------------------------------------------------------------------
extern "C" void kernel_launch(void* const* d_in, const int* in_sizes, int n_in,
                              void* d_out, int out_size, void* d_ws, size_t ws_size,
                              hipStream_t stream) {
    (void)in_sizes; (void)n_in; (void)out_size; (void)ws_size;

    const float* x      = (const float*)d_in[0];
    const float* gate_w = (const float*)d_in[1];
    const float* gate_b = (const float*)d_in[2];

    char* ws = (char*)d_ws;
    size_t off = 0;
    auto carve = [&](size_t bytes) -> char* {
        char* pp = ws + off;
        off = (off + bytes + 255) & ~(size_t)255;
        return pp;
    };

    int* top1 = (int*)carve(128 * sizeof(int));

    const int SLAB = 34 * 34 * 64;   // bf16 elems/sample, max geometry (layer1)
    const int XSLAB = 34 * 34 * 32;
    unsigned short* slab[4];
    for (int i = 0; i < 4; ++i)
        slab[i] = (unsigned short*)carve((size_t)128 * SLAB * sizeof(unsigned short));
    unsigned short* xslab =
        (unsigned short*)carve((size_t)128 * XSLAB * sizeof(unsigned short));

    // ---- bf16 weight conversion (per call; deterministic) ----
    struct WDesc { int idx, Cout, Cin, k2; };
    const WDesc wd[20] = {
        {  7,  64,   3, 9},  // 0 stem
        { 18,  64,  64, 9}, { 19,  64,  64, 9}, { 28,  64,  64, 9}, { 29,  64,  64, 9},
        { 42, 128,  64, 9}, { 43, 128, 128, 9}, { 44, 128,  64, 1},
        { 53, 128, 128, 9}, { 54, 128, 128, 9},
        { 67, 256, 128, 9}, { 68, 256, 256, 9}, { 69, 256, 128, 1},
        { 78, 256, 256, 9}, { 79, 256, 256, 9},
        { 92, 512, 256, 9}, { 93, 512, 512, 9}, { 94, 512, 256, 1},
        {103, 512, 512, 9}, {104, 512, 512, 9},
    };
    unsigned short* wbuf[20];
    for (int i = 0; i < 20; ++i) {
        int Cinp = wd[i].Cin < 32 ? 32 : wd[i].Cin;
        long total = (long)4 * wd[i].Cout * wd[i].k2 * Cinp;
        wbuf[i] = (unsigned short*)carve((size_t)total * sizeof(unsigned short));
        int blocks = (int)((total + 255) / 256);
        convert_w_kernel<<<blocks, 256, 0, stream>>>(
            (const float*)d_in[wd[i].idx], wbuf[i], total, wd[i].Cin, Cinp, wd[i].k2);
    }

    // ---- stem input conversion + gate ----
    {
        long total = 128L * XSLAB;
        convert_x_kernel<<<(int)((total + 255) / 256), 256, 0, stream>>>(x, xslab);
    }
    gate_kernel<<<128, 32, 0, stream>>>(x, gate_w, gate_b, top1);

    // ---- conv launcher: zero output slab (halo!) then fused conv ----
    const long SLAB16 = (128L * SLAB * 2) / 16;   // 16B granules per slab
    auto conv = [&](int wpos, int bnIdx,
                    const unsigned short* inB, int inStr, int Cinp, int WpIn,
                    const unsigned short* resB,
                    unsigned short* outB,
                    int Cout, int Hout, int Wout, int k2, int strd, int relu) {
        zero_kernel<<<(int)((SLAB16 + 255) / 256), 256, 0, stream>>>(
            (u32x4*)outB, SLAB16);
        int Npix = Hout * Wout;
        // wave tile: 32ch x (16*NSUB) px; pick arrangement so tiling is exact
        int nsub, wavesNLog;
        if (Npix >= 256)      { nsub = 2; wavesNLog = 2; } // block 64ch x 128px
        else if (Npix == 64)  { nsub = 2; wavesNLog = 1; } // block 128ch x 64px
        else                  { nsub = 1; wavesNLog = 0; } // block 256ch x 16px
        int wavesM = 8 >> wavesNLog;
        int hoff = (k2 == 9) ? 0 : 1;
        dim3 grid(Cout / (32 * wavesM),
                  Npix / (16 * nsub * (1 << wavesNLog)), 128);
        if (nsub == 2)
            conv_bn_kernel<2, 2><<<grid, 256, 0, stream>>>(
                wbuf[wpos],
                (const float*)d_in[bnIdx + 0], (const float*)d_in[bnIdx + 1],
                (const float*)d_in[bnIdx + 2], (const float*)d_in[bnIdx + 3],
                inB, inStr, resB, outB, SLAB, top1,
                Cinp, WpIn, Cout, Hout, Wout, Wout + 2,
                k2, strd, hoff, wavesNLog, relu);
        else
            conv_bn_kernel<2, 1><<<grid, 256, 0, stream>>>(
                wbuf[wpos],
                (const float*)d_in[bnIdx + 0], (const float*)d_in[bnIdx + 1],
                (const float*)d_in[bnIdx + 2], (const float*)d_in[bnIdx + 3],
                inB, inStr, resB, outB, SLAB, top1,
                Cinp, WpIn, Cout, Hout, Wout, Wout + 2,
                k2, strd, hoff, wavesNLog, relu);
    };

    const int S = SLAB;
    // stem: xslab(32ch,34) -> slab0 (64,32x32)
    conv(0,  3, xslab, XSLAB, 32, 34, nullptr, slab[0], 64, 32, 32, 9, 1, 1);
    // layer1
    conv(1, 10, slab[0], S, 64, 34, nullptr, slab[1], 64, 32, 32, 9, 1, 1);
    conv(2, 14, slab[1], S, 64, 34, slab[0], slab[2], 64, 32, 32, 9, 1, 1);
    conv(3, 20, slab[2], S, 64, 34, nullptr, slab[0], 64, 32, 32, 9, 1, 1);
    conv(4, 24, slab[0], S, 64, 34, slab[2], slab[1], 64, 32, 32, 9, 1, 1);
    // layer2
    conv(7, 38, slab[1], S, 64, 34, nullptr, slab[2], 128, 16, 16, 1, 2, 0);
    conv(5, 30, slab[1], S, 64, 34, nullptr, slab[0], 128, 16, 16, 9, 2, 1);
    conv(6, 34, slab[0], S, 128, 18, slab[2], slab[3], 128, 16, 16, 9, 1, 1);
    conv(8, 45, slab[3], S, 128, 18, nullptr, slab[0], 128, 16, 16, 9, 1, 1);
    conv(9, 49, slab[0], S, 128, 18, slab[3], slab[1], 128, 16, 16, 9, 1, 1);
    // layer3
    conv(12, 63, slab[1], S, 128, 18, nullptr, slab[2], 256, 8, 8, 1, 2, 0);
    conv(10, 55, slab[1], S, 128, 18, nullptr, slab[0], 256, 8, 8, 9, 2, 1);
    conv(11, 59, slab[0], S, 256, 10, slab[2], slab[3], 256, 8, 8, 9, 1, 1);
    conv(13, 70, slab[3], S, 256, 10, nullptr, slab[0], 256, 8, 8, 9, 1, 1);
    conv(14, 74, slab[0], S, 256, 10, slab[3], slab[1], 256, 8, 8, 9, 1, 1);
    // layer4
    conv(17, 88, slab[1], S, 256, 10, nullptr, slab[2], 512, 4, 4, 1, 2, 0);
    conv(15, 80, slab[1], S, 256, 10, nullptr, slab[0], 512, 4, 4, 9, 2, 1);
    conv(16, 84, slab[0], S, 512, 6, slab[2], slab[3], 512, 4, 4, 9, 1, 1);
    conv(18, 95, slab[3], S, 512, 6, nullptr, slab[0], 512, 4, 4, 9, 1, 1);
    conv(19, 99, slab[0], S, 512, 6, slab[3], slab[1], 512, 4, 4, 9, 1, 1);
    // head
    head_kernel<<<128, 256, 0, stream>>>(slab[1], SLAB,
                                         (const float*)d_in[9],
                                         (const float*)d_in[8],
                                         top1, (float*)d_out);
}
